// StyleGANGenerator_52226802319457
// MI455X (gfx1250) — compile-verified, matrix operand order checked by reading, not run on hardware
//
#include <hip/hip_runtime.h>
#include <hip/hip_bf16.h>

typedef __attribute__((ext_vector_type(16))) _Float16 v16h;
typedef __attribute__((ext_vector_type(8)))  float    v8f;

#define LRELU(v) ((v) > 0.f ? (v) : 0.2f * (v))

__device__ __forceinline__ float4 ld4(const float* p) { return *(const float4*)p; }

// K-offset within a 32-wide K chunk for the 16x16x32 f16 WMMA per-lane layout.
// lanes 0-15: h<8 -> K=h, h>=8 -> K=h+8 (16..23); lanes 16-31: +8.
__device__ __forceinline__ int koff(int h, int hi) {
    return (h < 8 ? h : h + 8) + 8 * hi;
}

#define LDSP  17  // pitch (floats) for dense B tile (scalar-staged)
#define LDSPA 20  // pitch (floats) for modconv B tile: 80B = 16B-aligned rows,
                  // lane halves differ by 640B == bank+32 -> conflict-free

// Convert two float4 pairs (x*s*msk) into am[off..off+7]
#define CVT8(am, off, xa, xb, sa, sb, msk)                                    \
    do {                                                                      \
        am[(off) + 0] = (_Float16)((xa).x * (sa).x * (msk));                  \
        am[(off) + 1] = (_Float16)((xa).y * (sa).y * (msk));                  \
        am[(off) + 2] = (_Float16)((xa).z * (sa).z * (msk));                  \
        am[(off) + 3] = (_Float16)((xa).w * (sa).w * (msk));                  \
        am[(off) + 4] = (_Float16)((xb).x * (sb).x * (msk));                  \
        am[(off) + 5] = (_Float16)((xb).y * (sb).y * (msk));                  \
        am[(off) + 6] = (_Float16)((xb).z * (sb).z * (msk));                  \
        am[(off) + 7] = (_Float16)((xb).w * (sb).w * (msk));                  \
    } while (0)

// ---------------- mapping network: one dense layer, M=4 rows, WMMA ----------
__global__ __launch_bounds__(32)
void dense_wmma(const float* __restrict__ in, const float* __restrict__ W,
                const float* __restrict__ bias, float* __restrict__ out,
                int K, int N, int Mrows) {
    __shared__ float Bs[32 * LDSP];
    int lane = threadIdx.x;
    int n0   = blockIdx.x * 16;
    int row  = lane & 15;
    int hi   = lane >> 4;
    int rr   = row < Mrows ? row : Mrows - 1;          // clamped A row
    float amask = row < Mrows ? 1.f : 0.f;
    v8f acc = {};
    for (int k0 = 0; k0 < K; k0 += 32) {
        // --- stage B tile [32 x 16] to LDS, coalesced b128 loads ---
        const float* wrow = W + (size_t)(k0 + lane) * N + n0;
        float4 w0 = ld4(wrow + 0), w1 = ld4(wrow + 4);
        float4 w2 = ld4(wrow + 8), w3 = ld4(wrow + 12);
        __syncthreads();
        float* bs = Bs + lane * LDSP;
        bs[0] = w0.x; bs[1] = w0.y; bs[2]  = w0.z; bs[3]  = w0.w;
        bs[4] = w1.x; bs[5] = w1.y; bs[6]  = w1.z; bs[7]  = w1.w;
        bs[8] = w2.x; bs[9] = w2.y; bs[10] = w2.z; bs[11] = w2.w;
        bs[12] = w3.x; bs[13] = w3.y; bs[14] = w3.z; bs[15] = w3.w;
        __syncthreads();

        // --- A: two contiguous 8-float runs, clamp+mask (no divergent loads) ---
        int base = 8 * hi;
        const float* xp = in + (size_t)rr * K + k0 + base;
        float4 xa = ld4(xp + 0), xb = ld4(xp + 4);
        float4 xc = ld4(xp + 16), xd = ld4(xp + 20);
        v16h a, b;
        a[0] = (_Float16)(xa.x * amask); a[1] = (_Float16)(xa.y * amask);
        a[2] = (_Float16)(xa.z * amask); a[3] = (_Float16)(xa.w * amask);
        a[4] = (_Float16)(xb.x * amask); a[5] = (_Float16)(xb.y * amask);
        a[6] = (_Float16)(xb.z * amask); a[7] = (_Float16)(xb.w * amask);
        a[8]  = (_Float16)(xc.x * amask); a[9]  = (_Float16)(xc.y * amask);
        a[10] = (_Float16)(xc.z * amask); a[11] = (_Float16)(xc.w * amask);
        a[12] = (_Float16)(xd.x * amask); a[13] = (_Float16)(xd.y * amask);
        a[14] = (_Float16)(xd.z * amask); a[15] = (_Float16)(xd.w * amask);
#pragma unroll
        for (int h = 0; h < 16; ++h)
            b[h] = (_Float16)Bs[koff(h, hi) * LDSP + row];
        acc = __builtin_amdgcn_wmma_f32_16x16x32_f16(false, a, false, b,
                                                     (short)0, acc, false, false);
    }
    if (lane < 16) {
        int n = n0 + lane;
#pragma unroll
        for (int r = 0; r < 4; ++r) {   // rows 0..3 (B=4) live in VGPR r, lanes 0-15
            if (r < Mrows) {
                float v = acc[r] + bias[n];
                out[r * N + n] = LRELU(v);
            }
        }
    }
}

// ---------------- tile const across batch ----------------
__global__ void tile_kernel(const float* __restrict__ c, float* __restrict__ o,
                            int n, int period) {
    int i = blockIdx.x * blockDim.x + threadIdx.x;
    if (i < n) o[i] = c[i % period];
}

// ---------------- bilinear 2x upsample (half-pixel centers) ----------------
__global__ void upsample_kernel(const float* __restrict__ in, float* __restrict__ out,
                                int H, int W, int C, int n) {
    int idx = blockIdx.x * blockDim.x + threadIdx.x;
    if (idx >= n) return;
    int H2 = H * 2, W2 = W * 2;
    int c = idx % C; int r = idx / C;
    int x = r % W2;  r /= W2;
    int y = r % H2;  int b = r / H2;
    float sy = (y + 0.5f) * 0.5f - 0.5f;
    float sx = (x + 0.5f) * 0.5f - 0.5f;
    float fy = floorf(sy), fx = floorf(sx);
    float wy = sy - fy,    wx = sx - fx;
    int y0 = (int)fy, x0 = (int)fx;
    int y0c = y0 < 0 ? 0 : (y0 > H - 1 ? H - 1 : y0);
    int y1c = (y0 + 1) < 0 ? 0 : ((y0 + 1) > H - 1 ? H - 1 : y0 + 1);
    int x0c = x0 < 0 ? 0 : (x0 > W - 1 ? W - 1 : x0);
    int x1c = (x0 + 1) < 0 ? 0 : ((x0 + 1) > W - 1 ? W - 1 : x0 + 1);
    const float* base = in + (size_t)b * H * W * C;
    float v00 = base[(y0c * W + x0c) * C + c];
    float v01 = base[(y0c * W + x1c) * C + c];
    float v10 = base[(y1c * W + x0c) * C + c];
    float v11 = base[(y1c * W + x1c) * C + c];
    out[idx] = (1.f - wy) * ((1.f - wx) * v00 + wx * v01) +
               wy * ((1.f - wx) * v10 + wx * v11);
}

// ---------------- style dense: s = w @ Ws + bs  (B=4, N=2f) ----------------
__global__ void style_kernel(const float* __restrict__ w, const float* __restrict__ Ws,
                             const float* __restrict__ bs, float* __restrict__ s, int N) {
    int idx = blockIdx.x * blockDim.x + threadIdx.x;
    if (idx >= 4 * N) return;
    int b = idx / N, n = idx % N;
    float acc = bs[n];
    for (int k = 0; k < 512; ++k) acc = fmaf(w[b * 512 + k], Ws[k * N + n], acc);
    s[idx] = acc;
}

// ---------------- K2[ci,f] = sum over 16 taps of kernel^2 ----------------
__global__ void k2_kernel(const float* __restrict__ k, float* __restrict__ k2,
                          int cin, int f) {
    int idx = blockIdx.x * blockDim.x + threadIdx.x;
    if (idx >= cin * f) return;
    int ci = idx / f, fo = idx % f;
    float acc = 0.f;
    for (int t = 0; t < 16; ++t) {
        float v = k[(t * cin + ci) * f + fo];
        acc = fmaf(v, v, acc);
    }
    k2[idx] = acc;
}

// ---------------- demod[b,f] = rsqrt(sum_ci s^2 * K2 + eps) ----------------
__global__ void demod_kernel(const float* __restrict__ s, const float* __restrict__ k2,
                             float* __restrict__ dm, int cin, int f) {
    int idx = blockIdx.x * blockDim.x + threadIdx.x;
    if (idx >= 4 * f) return;
    int b = idx / f, fo = idx % f;
    float acc = 1e-8f;
    for (int ci = 0; ci < cin; ++ci) {
        float sv = s[b * cin + ci];
        acc = fmaf(sv * sv, k2[ci * f + fo], acc);
    }
    dm[idx] = rsqrtf(acc);
}

// ---------------- modulated conv (batch-into-height faithful), WMMA --------
// X:[G=4*H2, W2, cin]  kern:[4,4,cin,f]  s:[4,cin]  dm:[4,f]
// noise/Y:[4,H2,W2,f].  Fuses lrelu + noise.
// B tile is staged to LDS with GLOBAL_LOAD_ASYNC_TO_LDS_B128 (ASYNCcnt path).
__global__ __launch_bounds__(32)
void modconv_wmma(const float* __restrict__ X, const float* __restrict__ kern,
                  const float* __restrict__ s, const float* __restrict__ dm,
                  const float* __restrict__ noise, float* __restrict__ Y,
                  int H2, int W2, int cin, int f) {
    __shared__ float Bs[32 * LDSPA];
    int lane = threadIdx.x;
    int G = 4 * H2;
    int tile_m = blockIdx.x;
    int n0 = blockIdx.y * 16;
    int row = lane & 15;
    int hi  = lane >> 4;
    int base = 8 * hi;
    int p = tile_m * 16 + row;           // output pixel this lane loads A rows for
    int g = p / W2, x = p % W2;

    // LDS byte address of this lane's staging row (flat addr low 32 bits == LDS offset)
    float* bs = Bs + lane * LDSPA;
    unsigned ldsdst = (unsigned)(unsigned long long)bs;

    v8f acc0 = {}, acc1 = {}, acc2 = {}, acc3 = {};

    for (int i = 0; i < 4; ++i) {
        for (int kx = 0; kx < 4; ++kx) {
            int sx = x + kx - 1;
            bool xok = (sx >= 0) && (sx < W2);
            int sxc = sx < 0 ? 0 : (sx >= W2 ? W2 - 1 : sx);
            const float* kbase = kern + (size_t)(i * 4 + kx) * cin * f;
            for (int k0 = 0; k0 < cin; k0 += 32) {
                // --- stage B tile [32 x 16] to LDS ---
                const float* krow = kbase + (size_t)(k0 + lane) * f + n0;
                // prior ds_load reads of Bs must have completed before overwrite
                asm volatile("s_wait_dscnt 0x0" ::: "memory");
                if (f - n0 >= 16) {
                    // 64B per lane, global->LDS direct (IOFFSET offsets both sides)
                    asm volatile(
                        "global_load_async_to_lds_b128 %0, %1, off\n\t"
                        "global_load_async_to_lds_b128 %0, %1, off offset:16\n\t"
                        "global_load_async_to_lds_b128 %0, %1, off offset:32\n\t"
                        "global_load_async_to_lds_b128 %0, %1, off offset:48"
                        :: "v"(ldsdst), "v"(krow) : "memory");
                    asm volatile("s_wait_asynccnt 0x0" ::: "memory");
                } else {                  // partial tile (f==8): masked scalar
#pragma unroll
                    for (int c = 0; c < 16; ++c)
                        bs[c] = (n0 + c < f) ? krow[c] : 0.f;
                }
                __syncthreads();
                v16h bm;
#pragma unroll
                for (int h = 0; h < 16; ++h)
                    bm[h] = (_Float16)Bs[koff(h, hi) * LDSPA + row];

#pragma unroll
                for (int bp = 0; bp < 4; ++bp) {
                    int t = bp * 4 + i - 7;
                    int sg = g + t;
                    int sgc = sg < 0 ? 0 : (sg >= G ? G - 1 : sg);
                    float msk = (xok && sg >= 0 && sg < G) ? 1.f : 0.f;
                    const float* xp = X + ((size_t)sgc * W2 + sxc) * cin + k0 + base;
                    const float* sp = s + bp * cin + k0 + base;
                    float4 xa = ld4(xp + 0),  xb = ld4(xp + 4);
                    float4 xc = ld4(xp + 16), xd = ld4(xp + 20);
                    float4 sa = ld4(sp + 0),  sb = ld4(sp + 4);
                    float4 sc = ld4(sp + 16), sd = ld4(sp + 20);
                    v16h am;
                    CVT8(am, 0, xa, xb, sa, sb, msk);
                    CVT8(am, 8, xc, xd, sc, sd, msk);
                    if (bp == 0)
                        acc0 = __builtin_amdgcn_wmma_f32_16x16x32_f16(false, am, false, bm,
                                                                      (short)0, acc0, false, false);
                    else if (bp == 1)
                        acc1 = __builtin_amdgcn_wmma_f32_16x16x32_f16(false, am, false, bm,
                                                                      (short)0, acc1, false, false);
                    else if (bp == 2)
                        acc2 = __builtin_amdgcn_wmma_f32_16x16x32_f16(false, am, false, bm,
                                                                      (short)0, acc2, false, false);
                    else
                        acc3 = __builtin_amdgcn_wmma_f32_16x16x32_f16(false, am, false, bm,
                                                                      (short)0, acc3, false, false);
                }
            }
        }
    }

    int n = n0 + row;
    if (n < f) {
        float d0 = dm[0 * f + n], d1 = dm[1 * f + n], d2 = dm[2 * f + n], d3 = dm[3 * f + n];
#pragma unroll
        for (int r = 0; r < 8; ++r) {
            int m = r + 8 * hi;
            int pp = tile_m * 16 + m;
            int gg = pp / W2, xx = pp % W2;
            int b = gg / H2, hh = gg % H2;
            float v = d0 * acc0[r] + d1 * acc1[r] + d2 * acc2[r] + d3 * acc3[r];
            v = LRELU(v);
            size_t oi = (((size_t)b * H2 + hh) * W2 + xx) * f + n;
            Y[oi] = v + noise[oi];
        }
    }
}

// ---------------- AdaIN: per (b,c) mean / rstd over H*W ----------------
__global__ void adain_stats(const float* __restrict__ X, float* __restrict__ mean,
                            float* __restrict__ rstd, int HW, int f) {
    __shared__ float ssum[256];
    __shared__ float ssq[256];
    int bc = blockIdx.x;          // b*f + c
    int b = bc / f, c = bc % f;
    int tid = threadIdx.x;
    float sum = 0.f, sq = 0.f;
    for (int p = tid; p < HW; p += 256) {
        float v = X[((size_t)b * HW + p) * f + c];
        sum += v; sq = fmaf(v, v, sq);
    }
    ssum[tid] = sum; ssq[tid] = sq;
    __syncthreads();
    for (int st = 128; st > 0; st >>= 1) {
        if (tid < st) { ssum[tid] += ssum[tid + st]; ssq[tid] += ssq[tid + st]; }
        __syncthreads();
    }
    if (tid == 0) {
        float m = ssum[0] / HW;
        mean[bc] = m;
        rstd[bc] = rsqrtf(ssq[0] / HW - m * m + 1e-5f);
    }
}

// ---------------- AdaIN apply, faithful stack/slice/reshape scramble --------
__global__ void adain_apply(float* __restrict__ X, const float* __restrict__ s,
                            const float* __restrict__ mean, const float* __restrict__ rstd,
                            int H2, int f, int n) {
    int idx = blockIdx.x * blockDim.x + threadIdx.x;
    if (idx >= n) return;
    int c = idx % f; int r = idx / f;
    int b = r / (H2 * H2);
    int half = f >> 1;
    int e = b >> 1;
    int bsrc = 2 * (b & 1) + (c >= half ? 1 : 0);
    int cm = c & (half - 1);    // f is a power of two
    int twof = 2 * f;
    float scale = s[bsrc * twof + e * f + cm];
    float shift = s[bsrc * twof + e * f + half + cm];
    int bc = b * f + c;
    float xn = (X[idx] - mean[bc]) * rstd[bc];
    X[idx] = scale * xn + shift;
}

// ---------------- final 3x3 conv, cin=8 -> 1, SAME pad ----------------
__global__ void final_conv(const float* __restrict__ in, const float* __restrict__ k,
                           float* __restrict__ out) {
    int idx = blockIdx.x * blockDim.x + threadIdx.x;
    if (idx >= 4 * 256 * 256) return;
    int x = idx % 256, y = (idx / 256) % 256, b = idx / (256 * 256);
    float acc = 0.f;
    for (int ky = 0; ky < 3; ++ky) {
        int yy = y + ky - 1;
        if (yy < 0 || yy >= 256) continue;
        for (int kx = 0; kx < 3; ++kx) {
            int xx = x + kx - 1;
            if (xx < 0 || xx >= 256) continue;
            const float* ip = in + (((size_t)b * 256 + yy) * 256 + xx) * 8;
            const float* kp = k + (ky * 3 + kx) * 8;
#pragma unroll
            for (int ci = 0; ci < 8; ++ci) acc = fmaf(ip[ci], kp[ci], acc);
        }
    }
    out[idx] = acc;
}

static const int FILT[6] = {256, 128, 64, 32, 16, 8};
static const int CIN_[6] = {512, 256, 128, 64, 32, 16};

extern "C" void kernel_launch(void* const* d_in, const int* in_sizes, int n_in,
                              void* d_out, int out_size, void* d_ws, size_t ws_size,
                              hipStream_t stream) {
    const float* z1 = (const float*)d_in[0];
    const float* map_ws[8]; const float* map_bs[8];
    for (int i = 0; i < 8; ++i) { map_ws[i] = (const float*)d_in[1 + i];
                                  map_bs[i] = (const float*)d_in[9 + i]; }
    const float* cst = (const float*)d_in[17];
    const float* conv_ks[6]; const float* style_ws[6]; const float* style_bs[6];
    const float* noises[6];
    for (int i = 0; i < 6; ++i) { conv_ks[i]  = (const float*)d_in[18 + i];
                                  style_ws[i] = (const float*)d_in[24 + i];
                                  style_bs[i] = (const float*)d_in[30 + i]; }
    const float* final_k = (const float*)d_in[36];
    for (int i = 0; i < 6; ++i) noises[i] = (const float*)d_in[37 + i];

    float* ws = (float*)d_ws;
    size_t off = 0;
    const size_t ACT = (size_t)4 * 256 * 256 * 16;  // 4.19M floats, max activation
    float* bufA   = ws + off; off += ACT;
    float* bufB   = ws + off; off += ACT;
    float* w0     = ws + off; off += 2048;
    float* w1     = ws + off; off += 2048;
    float* sbuf   = ws + off; off += 4 * 512;
    float* k2buf  = ws + off; off += 512 * 256;
    float* dmbuf  = ws + off; off += 4 * 256;
    float* mbuf   = ws + off; off += 4 * 256;
    float* rbuf   = ws + off; off += 4 * 256;
    (void)off; (void)ws_size; (void)in_sizes; (void)n_in; (void)out_size;

    // ---- mapping network: 8 dense+lrelu layers via WMMA ----
    const float* cur = z1;
    float* nxt = w0;
    for (int l = 0; l < 8; ++l) {
        dense_wmma<<<dim3(512 / 16), 32, 0, stream>>>(cur, map_ws[l], map_bs[l], nxt,
                                                      512, 512, 4);
        cur = nxt;
        nxt = (nxt == w0) ? w1 : w0;
    }
    const float* wlat = cur;

    // ---- const tile -> bufA [4,4,4,512] ----
    tile_kernel<<<(32768 + 255) / 256, 256, 0, stream>>>(cst, bufA, 32768, 8192);

    int H = 4;
    float* P = bufA;  // current activation
    float* Q = bufB;  // upsample scratch
    for (int i = 0; i < 6; ++i) {
        int H2 = H * 2, f = FILT[i], cin = CIN_[i];
        int nup = 4 * H2 * H2 * cin;
        upsample_kernel<<<(nup + 255) / 256, 256, 0, stream>>>(P, Q, H, H, cin, nup);

        int N = 2 * f;
        style_kernel<<<(4 * N + 255) / 256, 256, 0, stream>>>(wlat, style_ws[i],
                                                              style_bs[i], sbuf, N);
        k2_kernel<<<(cin * f + 255) / 256, 256, 0, stream>>>(conv_ks[i], k2buf, cin, f);
        demod_kernel<<<(4 * f + 255) / 256, 256, 0, stream>>>(sbuf, k2buf, dmbuf, cin, f);

        int tiles_m = (4 * H2 * H2) / 16;
        int tiles_n = (f + 15) / 16;
        modconv_wmma<<<dim3(tiles_m, tiles_n), 32, 0, stream>>>(Q, conv_ks[i], sbuf,
                                                                dmbuf, noises[i], P,
                                                                H2, H2, cin, f);

        adain_stats<<<4 * f, 256, 0, stream>>>(P, mbuf, rbuf, H2 * H2, f);
        int nel = 4 * H2 * H2 * f;
        adain_apply<<<(nel + 255) / 256, 256, 0, stream>>>(P, sbuf, mbuf, rbuf, H2, f, nel);

        H = H2;
    }

    // ---- final 3x3 conv -> d_out [4,256,256,1] ----
    final_conv<<<(4 * 256 * 256 + 255) / 256, 256, 0, stream>>>(P, final_k, (float*)d_out);
}
